// SelfAttention_87651692577154
// MI455X (gfx1250) — matrix operand order repaired and checked by reading, not compiled, not run on hardware
//
#include <hip/hip_runtime.h>
#include <stdint.h>

// ---------------------------------------------------------------------------
// Self-attention for MI455X (gfx1250, wave32, WMMA + async-LDS).
//   B=8, S=2048, E=1024.
//   1) proj_kernel (x3): Q/K/V = x W^T + b, f32 loads -> f16 convert in regs,
//      register-double-buffered LDS, 256x128 block tile, 64x64 per wave,
//      v_wmma_f32_16x16x32_f16.  V stored transposed per batch [b][e][s].
//   2) gemm_f16_kernel (scores, out): f16 inputs staged with
//      global_load_async_to_lds_b128 (ASYNCcnt-tracked, LDS double buffer).
//   3) softmax_rows_kernel: row softmax, writes f16 weights in place.
// Workspace (bytes): Qh 33.5M | Kh 33.5M | Vt 33.5M | scores f32 [8][S][S] 134M
//   total ~235 MB.
// ---------------------------------------------------------------------------

typedef __attribute__((ext_vector_type(16))) _Float16 v16h;
typedef __attribute__((ext_vector_type(8)))  _Float16 v8h;
typedef __attribute__((ext_vector_type(8)))  float    v8f;

#define LSTR 40  // LDS row stride in halves: 32 payload + 8 pad (80B, 16B-aligned)

// --- async copy: 32 contiguous bytes global -> LDS (2 x b128, ASYNCcnt) ----
__device__ __forceinline__ uint32_t lds_lo32(const void* p) {
  // generic pointers into the LDS aperture carry the LDS byte offset in the
  // low 32 bits (ISA 10.2: flat->LDS truncates to addr[31:0]).
  return (uint32_t)(uintptr_t)p;
}

__device__ __forceinline__ void async_copy32(uint32_t lds, const _Float16* g) {
  asm volatile(
      "global_load_async_to_lds_b128 %0, %1, off\n\t"
      "global_load_async_to_lds_b128 %0, %1, off offset:16"
      :: "v"(lds), "v"(g) : "memory");
}

__device__ __forceinline__ void wait_async6() {
#if __has_builtin(__builtin_amdgcn_s_wait_asynccnt)
  __builtin_amdgcn_s_wait_asynccnt(6);
#else
  asm volatile("s_wait_asynccnt 0x6" ::: "memory");
#endif
}
__device__ __forceinline__ void wait_async0() {
#if __has_builtin(__builtin_amdgcn_s_wait_asynccnt)
  __builtin_amdgcn_s_wait_asynccnt(0);
#else
  asm volatile("s_wait_asynccnt 0x0" ::: "memory");
#endif
}

// --- f32 staging: 16 floats -> regs -> f16 -> 32B LDS store -----------------
__device__ __forceinline__ void load16f(const float* __restrict__ s, float4 (&r)[4]) {
  const float4* s4 = reinterpret_cast<const float4*>(s);
  r[0] = s4[0]; r[1] = s4[1]; r[2] = s4[2]; r[3] = s4[3];
}

__device__ __forceinline__ void cvtstore16(const float4 (&r)[4], _Float16* __restrict__ d) {
  v8h h0, h1;
  h0[0] = (_Float16)r[0].x; h0[1] = (_Float16)r[0].y; h0[2] = (_Float16)r[0].z; h0[3] = (_Float16)r[0].w;
  h0[4] = (_Float16)r[1].x; h0[5] = (_Float16)r[1].y; h0[6] = (_Float16)r[1].z; h0[7] = (_Float16)r[1].w;
  h1[0] = (_Float16)r[2].x; h1[1] = (_Float16)r[2].y; h1[2] = (_Float16)r[2].z; h1[3] = (_Float16)r[2].w;
  h1[4] = (_Float16)r[3].x; h1[5] = (_Float16)r[3].y; h1[6] = (_Float16)r[3].z; h1[7] = (_Float16)r[3].w;
  *reinterpret_cast<v8h*>(d)     = h0;
  *reinterpret_cast<v8h*>(d + 8) = h1;
}

// --- fragment loads from LDS per CDNA5 WMMA VGPR layouts --------------------
// A (16x32 f16 MxK): lane&15=M; lanes 0-15: K 0-7 & 16-23; lanes 16-31: +8.
__device__ __forceinline__ v16h loadAfrag(const _Float16* lA, int row0, int lane) {
  const int m  = row0 + (lane & 15);
  const int kb = (lane >> 4) * 8;
  const _Float16* p = lA + m * LSTR + kb;
  union { v16h v; v8h h[2]; } u;
  u.h[0] = *reinterpret_cast<const v8h*>(p);
  u.h[1] = *reinterpret_cast<const v8h*>(p + 16);
  return u.v;
}
// B (32x16 f16 KxN, stored [n][k]): lane&15=N; lanes 0-15: K 0-7 & 16-23; lanes 16-31: +8.
__device__ __forceinline__ v16h loadBfrag(const _Float16* lB, int col0, int lane) {
  const int n  = col0 + (lane & 15);
  const int kb = (lane >> 4) * 8;
  const _Float16* p = lB + n * LSTR + kb;
  union { v16h v; v8h h[2]; } u;
  u.h[0] = *reinterpret_cast<const v8h*>(p);
  u.h[1] = *reinterpret_cast<const v8h*>(p + 16);
  return u.v;
}

// --- one 64x64 wave tile worth of WMMA over a 32-wide K stage ---------------
__device__ __forceinline__ void wmma_tile(const _Float16* lA, const _Float16* lB,
                                          int wm, int wn, int lane, v8f (&acc)[4][4]) {
  v16h aV[4], bV[4];
#pragma unroll
  for (int i = 0; i < 4; ++i) aV[i] = loadAfrag(lA, wm * 64 + i * 16, lane);
#pragma unroll
  for (int i = 0; i < 4; ++i) bV[i] = loadBfrag(lB, wn * 64 + i * 16, lane);
#pragma unroll
  for (int mi = 0; mi < 4; ++mi)
#pragma unroll
    for (int ni = 0; ni < 4; ++ni)
      acc[mi][ni] = __builtin_amdgcn_wmma_f32_16x16x32_f16(
          false, aV[mi], false, bV[ni], (short)0, acc[mi][ni], false, false);
}

// ---------------------------------------------------------------------------
// QKV projection: C[16384,1024](f16) = A[16384,1024](f32) x B[1024,1024]^T + bias.
// 256x128 block tile, 8 waves (4x2), 64x64 per wave, K=1024 stepped by 32.
// Register-staged double buffer: loads for stage k+1 overlap WMMA of stage k.
// TRANS_V: store per-batch transposed [b][e][s] (b = row>>11, s = row&2047).
// ---------------------------------------------------------------------------
template <bool TRANS_V>
__global__ __launch_bounds__(256)
void proj_kernel(const float* __restrict__ A, const float* __restrict__ Bm,
                 const float* __restrict__ bias, _Float16* __restrict__ C) {
  __shared__ alignas(16) _Float16 lA[2][256 * LSTR];
  __shared__ alignas(16) _Float16 lB[2][128 * LSTR];

  const int t = threadIdx.x, lane = t & 31, w = t >> 5;
  const int wm = w >> 1, wn = w & 1;
  const int tileM = blockIdx.y * 256, tileN = blockIdx.x * 128;
  const int cr = t >> 1, cc = (t & 1) * 16;

  const float* aP0 = A + (size_t)(tileM + cr) * 1024 + cc;
  const float* aP1 = A + (size_t)(tileM + 128 + cr) * 1024 + cc;
  const float* bP  = Bm + (size_t)(tileN + cr) * 1024 + cc;

  float4 ra0[4], ra1[4], rb[4];
  load16f(aP0, ra0); load16f(aP1, ra1); load16f(bP, rb);

  v8f acc[4][4] = {};

  for (int k0 = 0; k0 < 1024; k0 += 32) {
    const int buf = (k0 >> 5) & 1;
    cvtstore16(ra0, &lA[buf][cr * LSTR + cc]);
    cvtstore16(ra1, &lA[buf][(128 + cr) * LSTR + cc]);
    cvtstore16(rb,  &lB[buf][cr * LSTR + cc]);
    if (k0 + 32 < 1024) {
      aP0 += 32; aP1 += 32; bP += 32;
      load16f(aP0, ra0); load16f(aP1, ra1); load16f(bP, rb);
      __builtin_prefetch(aP0 + 32, 0, 3);
      __builtin_prefetch(bP + 32, 0, 3);
    }
    __syncthreads();
    wmma_tile(lA[buf], lB[buf], wm, wn, lane, acc);
    __syncthreads();
  }

  const int nl = lane & 15, hi = lane >> 4;
#pragma unroll
  for (int mi = 0; mi < 4; ++mi) {
#pragma unroll
    for (int ni = 0; ni < 4; ++ni) {
      const int gn = tileN + wn * 64 + ni * 16 + nl;
      const float bval = bias[gn];
#pragma unroll
      for (int v = 0; v < 8; ++v) {
        const int gm = tileM + wm * 64 + mi * 16 + hi * 8 + v;
        const float r = acc[mi][ni][v] + bval;
        if (TRANS_V) {
          const int b = gm >> 11, sq = gm & 2047;
          C[((size_t)b * 1024 + gn) * 2048 + sq] = (_Float16)r;
        } else {
          C[(size_t)gm * 1024 + gn] = (_Float16)r;
        }
      }
    }
  }
}

// ---------------------------------------------------------------------------
// f16 GEMM: C[M,N](f32) = alpha * A[M,K] x B[N,K]^T, batched over blockIdx.z.
// Staging via global_load_async_to_lds_b128 into a double-buffered LDS tile.
// Per wave per stage: 6 async ops in flight for next stage -> wait <= 6.
// ---------------------------------------------------------------------------
__global__ __launch_bounds__(256)
void gemm_f16_kernel(const _Float16* __restrict__ A, int lda, size_t aBatch,
                     const _Float16* __restrict__ Bm, int ldb, size_t bBatch,
                     float* __restrict__ C, int ldc, size_t cBatch,
                     float alpha, int K) {
  __shared__ alignas(16) _Float16 lA[2][256 * LSTR];
  __shared__ alignas(16) _Float16 lB[2][128 * LSTR];

  const int z = blockIdx.z;
  A  += (size_t)z * aBatch;
  Bm += (size_t)z * bBatch;
  C  += (size_t)z * cBatch;

  const int t = threadIdx.x, lane = t & 31, w = t >> 5;
  const int wm = w >> 1, wn = w & 1;
  const int tileM = blockIdx.y * 256, tileN = blockIdx.x * 128;
  const int cr = t >> 1, cc = (t & 1) * 16;

  const _Float16* gA0 = A + (size_t)(tileM + cr) * lda + cc;
  const _Float16* gA1 = A + (size_t)(tileM + 128 + cr) * lda + cc;
  const _Float16* gB  = Bm + (size_t)(tileN + cr) * ldb + cc;

  uint32_t la0[2], la1[2], lb[2];
#pragma unroll
  for (int b = 0; b < 2; ++b) {
    la0[b] = lds_lo32(&lA[b][cr * LSTR + cc]);
    la1[b] = lds_lo32(&lA[b][(128 + cr) * LSTR + cc]);
    lb[b]  = lds_lo32(&lB[b][cr * LSTR + cc]);
  }

  // prologue: fill both buffers (ASYNCcnt += 6 per stage per wave)
  async_copy32(la0[0], gA0);
  async_copy32(la1[0], gA1);
  async_copy32(lb[0],  gB);
  if (32 < K) {
    async_copy32(la0[1], gA0 + 32);
    async_copy32(la1[1], gA1 + 32);
    async_copy32(lb[1],  gB + 32);
  }

  v8f acc[4][4] = {};

  for (int k0 = 0; k0 < K; k0 += 32) {
    const int buf = (k0 >> 5) & 1;
    if (k0 + 32 < K) wait_async6();  // next stage still in flight; cur done
    else             wait_async0();  // last stage: drain
    __syncthreads();                 // every wave's portion of cur complete
    wmma_tile(lA[buf], lB[buf], wm, wn, lane, acc);
    __syncthreads();                 // all reads of cur done before refill
    if (k0 + 64 < K) {
      async_copy32(la0[buf], gA0 + k0 + 64);
      async_copy32(la1[buf], gA1 + k0 + 64);
      async_copy32(lb[buf],  gB + k0 + 64);
    }
  }

  const int nl = lane & 15, hi = lane >> 4;
#pragma unroll
  for (int mi = 0; mi < 4; ++mi) {
#pragma unroll
    for (int ni = 0; ni < 4; ++ni) {
      const int gn = tileN + wn * 64 + ni * 16 + nl;
#pragma unroll
      for (int v = 0; v < 8; ++v) {
        const int gm = tileM + wm * 64 + mi * 16 + hi * 8 + v;
        C[(size_t)gm * ldc + gn] = acc[mi][ni][v] * alpha;
      }
    }
  }
}

// ---------------------------------------------------------------------------
// Row softmax over 2048 f32 scores; writes f16 weights in place over the
// same row.  All reads complete before the first barrier -> aliased f16
// stores after the reductions are race-free.
// ---------------------------------------------------------------------------
__global__ __launch_bounds__(256)
void softmax_rows_kernel(float* __restrict__ scores) {
  const int row = blockIdx.x;
  float* rp = scores + (size_t)row * 2048;
  const int t = threadIdx.x;

  float v[8];
#pragma unroll
  for (int i = 0; i < 8; ++i) v[i] = rp[t + i * 256];

  float m = -1e30f;
#pragma unroll
  for (int i = 0; i < 8; ++i) m = fmaxf(m, v[i]);
#pragma unroll
  for (int off = 16; off; off >>= 1) m = fmaxf(m, __shfl_xor(m, off, 32));
  __shared__ float redm[8];
  if ((t & 31) == 0) redm[t >> 5] = m;
  __syncthreads();
#pragma unroll
  for (int i = 0; i < 8; ++i) m = fmaxf(m, redm[i]);

  float s = 0.0f;
#pragma unroll
  for (int i = 0; i < 8; ++i) {
    v[i] = exp2f((v[i] - m) * 1.44269504f);
    s += v[i];
  }
#pragma unroll
  for (int off = 16; off; off >>= 1) s += __shfl_xor(s, off, 32);
  __shared__ float reds[8];
  if ((t & 31) == 0) reds[t >> 5] = s;
  __syncthreads();
  s = 0.0f;
#pragma unroll
  for (int i = 0; i < 8; ++i) s += reds[i];
  const float inv = 1.0f / s;

  _Float16* wp = reinterpret_cast<_Float16*>(rp);
#pragma unroll
  for (int i = 0; i < 8; ++i) wp[t + i * 256] = (_Float16)(v[i] * inv);
}

// ---------------------------------------------------------------------------
extern "C" void kernel_launch(void* const* d_in, const int* in_sizes, int n_in,
                              void* d_out, int out_size, void* d_ws, size_t ws_size,
                              hipStream_t stream) {
  (void)in_sizes; (void)n_in; (void)out_size; (void)ws_size;

  const float* x  = (const float*)d_in[0];
  const float* Wq = (const float*)d_in[1];
  const float* bq = (const float*)d_in[2];
  const float* Wk = (const float*)d_in[3];
  const float* bk = (const float*)d_in[4];
  const float* Wv = (const float*)d_in[5];
  const float* bv = (const float*)d_in[6];
  float* out = (float*)d_out;

  const int S = 2048, E = 1024, Bn = 8;
  const size_t rows = (size_t)Bn * S;  // 16384

  _Float16* Qh = (_Float16*)d_ws;
  _Float16* Kh = Qh + rows * E;
  _Float16* Vt = Kh + rows * E;  // [b][e][s]
  float* scores = (float*)((char*)d_ws + 3 * rows * E * sizeof(_Float16));  // [b][s][s]

  dim3 blk(256);

  // 1) QKV projections
  dim3 gProj(E / 128, (unsigned)(rows / 256));  // (8, 64)
  proj_kernel<false><<<gProj, blk, 0, stream>>>(x, Wq, bq, Qh);
  proj_kernel<false><<<gProj, blk, 0, stream>>>(x, Wk, bk, Kh);
  proj_kernel<true ><<<gProj, blk, 0, stream>>>(x, Wv, bv, Vt);

  // 2) scores = Q K^T * 1/sqrt(E), all batches in one launch
  dim3 gS(S / 128, S / 256, Bn);  // (16, 8, 8)
  gemm_f16_kernel<<<gS, blk, 0, stream>>>(
      Qh, E, (size_t)S * E, Kh, E, (size_t)S * E,
      scores, S, (size_t)S * S, 0.03125f, E);

  // 3) row softmax (f16 weights in place)
  softmax_rows_kernel<<<(unsigned)rows, blk, 0, stream>>>(scores);

  // 4) out = P V  (A rows are f16 packed in f32 score rows: lda = 2S halves)
  dim3 gO(E / 128, S / 256, Bn);  // (8, 8, 8)
  gemm_f16_kernel<<<gO, blk, 0, stream>>>(
      (const _Float16*)scores, 2 * S, (size_t)S * S * 2, Vt, S, (size_t)E * S,
      out, E, (size_t)S * E, 1.0f, S);
}
